// DUVRec_39187281608944
// MI455X (gfx1250) — matrix operand order repaired and checked by reading, compiled, not run on hardware
//
#include <hip/hip_runtime.h>
#include <hip/hip_bf16.h>
#include <math.h>

#define TEMP_ 0.2f

typedef __bf16 bf16;
typedef __attribute__((ext_vector_type(16))) __bf16  v16bf;
typedef __attribute__((ext_vector_type(8)))  float   v8f;

// PE_TABLE[t][d] computed on the fly (module-level constant in the reference)
__device__ __forceinline__ float pe_val(int t, int d){
  float ang = (float)t * __expf(-(float)(d & ~1) * (9.210340371976184f / 128.0f));
  return (d & 1) ? __cosf(ang) : __sinf(ang);
}

__device__ __forceinline__ v8f wmma_bf16(v16bf a, v16bf b, v8f c){
  return __builtin_amdgcn_wmma_f32_16x16x32_bf16(false, a, false, b, (short)0, c, false, false);
}

// ---- wave32 fragment builders (contiguous k = 16*(lane/16) + e, same in A and B) ----
// A 16x32 tile from row-major [m][k] LDS; one contiguous 32B run per lane.
__device__ __forceinline__ v16bf load_a(const bf16* p, int r0, int c0, int stride){
  int l = threadIdx.x & 31;
  const bf16* base = p + (r0 + (l & 15)) * stride + c0 + ((l >> 4) << 4);
  v16bf q;
  #pragma unroll
  for (int e = 0; e < 16; ++e) q[e] = base[e];
  return q;
}
// B 32x16 tile where LDS stores B "column-major": p[n][k], K contiguous.
// One 32B contiguous run per lane -> ds_load_b128 x2.
__device__ __forceinline__ v16bf load_b_nk(const bf16* p, int k0, int n0, int stride){
  int l = threadIdx.x & 31;
  const bf16* base = p + (n0 + (l & 15)) * stride + k0 + ((l >> 4) << 4);
  v16bf q;
  #pragma unroll
  for (int e = 0; e < 16; ++e) q[e] = base[e];
  return q;
}
// B fragment straight from a global row-major f32 weight matrix (cvt in flight).
__device__ __forceinline__ v16bf load_b_g(const float* g, int k0, int n0, int stride, int ncols){
  int l = threadIdx.x & 31;
  int n = n0 + (l & 15);
  const float* base = g + (k0 + ((l >> 4) << 4)) * stride + n;
  bool ok = n < ncols;
  v16bf q;
  #pragma unroll
  for (int e = 0; e < 16; ++e) q[e] = ok ? (bf16)base[e * stride] : (bf16)0.0f;
  return q;
}

__global__ void duv_zero4(float* o){ if (threadIdx.x < 4) o[threadIdx.x] = 0.f; }

// One workgroup (8 wave32s) per batch element; fully fused in ~122 KB LDS.
__global__ __launch_bounds__(256) void duv_block_kernel(
    const int* __restrict__ history, const int* __restrict__ timestp,
    const float* __restrict__ adj,   const int* __restrict__ pos_id,
    const int* __restrict__ neg_id,  const int* __restrict__ stp,
    const float* __restrict__ emb,   const float* __restrict__ W1,
    const float* __restrict__ b1,    const float* __restrict__ W2,
    const float* __restrict__ b2,    float* __restrict__ out)
{
  __shared__ bf16  xT_bf[128*64];    // x transposed [d][n]     16 KB (B operand M1,M4)
  __shared__ bf16  cand_bf[64*128];  // cand [c][d]             16 KB (A operand M5)
  __shared__ bf16  adj_bf[64*64];    // adj (pad) [r][c]         8 KB (A operand M1)
  __shared__ bf16  ax_bf[64*128];    // adj@x [m][k]            16 KB (A operand M2)
  __shared__ bf16  h_bf[64*256];     // relu(ax@W1+b1)          32 KB (A operand M3)
  __shared__ float lg[64*32];        // logits scratch           8 KB
  __shared__ bf16  sT_bf[32*64];     // s^T [a][n]               4 KB (A operand M4)
  __shared__ bf16  xp_bf[32*128];    // x_parent [a][d]          8 KB (B operand M5)
  __shared__ bf16  xpT_bf[128*32];   // x_parent^T [d][a]        8 KB (B operand M6)
  __shared__ bf16  attn_bf[64*32];   // attn [c][a]              4 KB (A operand M6)
  __shared__ float uI[128], uF[128];
  __shared__ float scores[64], scoresI[64], scoresF[64];
  __shared__ float red[2];           // [0]=reg sum, [1]=sum s*log s

  const int b   = blockIdx.x;
  const int tid = threadIdx.x;
  const int wid = tid >> 5;

  // ---- zero accumulators ----
  for (int i = tid; i < 128; i += 256){ uI[i] = 0.f; uF[i] = 0.f; }
  if (tid < 64) scores[tid] = 0.f;
  if (tid < 2)  red[tid] = 0.f;
  __syncthreads();

  // ---- gather x / cand / adj; accumulate reg sum and user_emb_I ----
  // stride-256 loop => d = tid & 127 is loop-invariant per thread
  float regAcc = 0.f, uIsum = 0.f;
  const int t_c  = stp[b];
  const int dfix = tid & 127;
  for (int i = tid; i < 64*128; i += 256){
    int n = i >> 7, d = i & 127;
    float xv = 0.f, cv = 0.f;
    if (n < 50){
      int hid = history[b*50+n];
      int tt  = timestp[b*50+n];
      xv = emb[hid*128+d] + pe_val(tt, d);
      regAcc += xv*xv;
      uIsum  += xv;
      int cid = (n == 0) ? pos_id[b] : neg_id[b*49 + n - 1];
      cv = emb[cid*128+d] + pe_val(t_c, d);
      regAcc += cv*cv;                              // pos + neg rows = all cand
    }
    xT_bf[d*64 + n]   = (bf16)xv;
    cand_bf[n*128 + d] = (bf16)cv;
  }
  atomicAdd(&uI[dfix], uIsum);
  for (int i = tid; i < 64*64; i += 256){
    int r = i >> 6, c = i & 63;
    float v = (r < 50 && c < 50) ? adj[b*2500 + r*50 + c] : 0.f;
    adj_bf[i] = (bf16)v;
  }
  atomicAdd(&red[0], regAcc);
  __syncthreads();

  // ---- M1: ax = adj @ x   (64x64 @ 64x128) ----
  for (int t = wid; t < 32; t += 8){
    int mt = t >> 3, nt = t & 7;
    v8f acc = {};
    for (int kt = 0; kt < 2; ++kt)
      acc = wmma_bf16(load_a(adj_bf, mt*16, kt*32, 64),
                      load_b_nk(xT_bf, kt*32, nt*16, 64), acc);
    int l = tid & 31, col = nt*16 + (l & 15), r0 = mt*16 + ((l>>4)<<3);
    #pragma unroll
    for (int i = 0; i < 8; ++i) ax_bf[(r0+i)*128 + col] = (bf16)acc[i];
  }
  __syncthreads();

  // ---- M2: h = relu(ax @ W1 + b1)   (64x128 @ 128x256) ----
  for (int t = wid; t < 64; t += 8){
    int mt = t >> 4, nt = t & 15;
    v8f acc = {};
    for (int kt = 0; kt < 4; ++kt)
      acc = wmma_bf16(load_a(ax_bf, mt*16, kt*32, 128),
                      load_b_g(W1,  kt*32, nt*16, 256, 256), acc);
    int l = tid & 31, col = nt*16 + (l & 15), r0 = mt*16 + ((l>>4)<<3);
    float bias = b1[col];
    #pragma unroll
    for (int i = 0; i < 8; ++i){
      float v = acc[i] + bias;
      h_bf[(r0+i)*256 + col] = (bf16)(v > 0.f ? v : 0.f);
    }
  }
  __syncthreads();

  // ---- M3: logits = h @ W2 + b2   (64x256 @ 256x20, pad N->32) ----
  {
    int mt = wid >> 1, nt = wid & 1;                  // exactly 8 tiles
    v8f acc = {};
    for (int kt = 0; kt < 8; ++kt)
      acc = wmma_bf16(load_a(h_bf, mt*16, kt*32, 256),
                      load_b_g(W2, kt*32, nt*16, 20, 20), acc);
    int l = tid & 31, col = nt*16 + (l & 15), r0 = mt*16 + ((l>>4)<<3);
    float bias = (col < 20) ? b2[col] : 0.f;
    #pragma unroll
    for (int i = 0; i < 8; ++i) lg[(r0+i)*32 + col] = acc[i] + bias;
  }
  __syncthreads();

  // ---- softmax -> s^T (zero pad rows n>=50, a>=20), entropy ----
  if (tid < 64){
    int n = tid;
    float ent = 0.f;
    if (n < 50){
      float m = -1e30f;
      for (int a = 0; a < 20; ++a) m = fmaxf(m, lg[n*32+a]);
      float e[20], sum = 0.f;
      for (int a = 0; a < 20; ++a){ e[a] = __expf(lg[n*32+a] - m); sum += e[a]; }
      float inv = 1.f / sum;
      for (int a = 0; a < 32; ++a){
        if (a < 20){
          float sv = e[a] * inv;
          sT_bf[a*64+n] = (bf16)sv;
          if (sv > 0.f) ent += sv * __logf(sv);
        } else sT_bf[a*64+n] = (bf16)0.0f;
      }
    } else {
      for (int a = 0; a < 32; ++a) sT_bf[a*64+n] = (bf16)0.0f;
    }
    atomicAdd(&red[1], ent);
  }
  __syncthreads();

  // ---- M4: x_parent = s^T @ x (32x64 @ 64x128); both orientations + user_emb_F ----
  for (int t = wid; t < 16; t += 8){
    int mt = t >> 3, nt = t & 7;
    v8f acc = {};
    for (int kt = 0; kt < 2; ++kt)
      acc = wmma_bf16(load_a(sT_bf, mt*16, kt*32, 64),
                      load_b_nk(xT_bf, kt*32, nt*16, 64), acc);
    int l = tid & 31, col = nt*16 + (l & 15), r0 = mt*16 + ((l>>4)<<3);
    #pragma unroll
    for (int i = 0; i < 8; ++i){
      int row = r0 + i;
      bf16 hv = (bf16)acc[i];
      xp_bf [row*128 + col] = hv;
      xpT_bf[col*32  + row] = hv;
      if (row < 20) atomicAdd(&uF[col], acc[i]);
    }
  }
  __syncthreads();

  // ---- M5: attn logits = cand @ x_parent^T / sqrt(128)  (64x128 @ 128x32) ----
  {
    int mt = wid >> 1, nt = wid & 1;                  // 8 tiles
    v8f acc = {};
    for (int kt = 0; kt < 4; ++kt)
      acc = wmma_bf16(load_a(cand_bf, mt*16, kt*32, 128),
                      load_b_nk(xp_bf, kt*32, nt*16, 128), acc);   // B[k=d][n=a]=xp[a][d]
    int l = tid & 31, col = nt*16 + (l & 15), r0 = mt*16 + ((l>>4)<<3);
    #pragma unroll
    for (int i = 0; i < 8; ++i) lg[(r0+i)*32 + col] = acc[i] * 0.08838834764831845f;
  }
  __syncthreads();

  // ---- softmax over a<20 -> attn ----
  if (tid < 64){
    int c = tid;
    float m = -1e30f;
    for (int a = 0; a < 20; ++a) m = fmaxf(m, lg[c*32+a]);
    float e[20], sum = 0.f;
    for (int a = 0; a < 20; ++a){ e[a] = __expf(lg[c*32+a] - m); sum += e[a]; }
    float inv = 1.f / sum;
    for (int a = 0; a < 32; ++a)
      attn_bf[c*32+a] = (a < 20) ? (bf16)(e[a]*inv) : (bf16)0.0f;
  }
  __syncthreads();

  // ---- M6: user4cand = attn @ x_parent (64x32 @ 32x128); fuse row-dot with cand ----
  for (int t = wid; t < 32; t += 8){
    int mt = t >> 3, nt = t & 7;
    v8f acc = wmma_bf16(load_a(attn_bf, mt*16, 0, 32),
                        load_b_nk(xpT_bf, 0, nt*16, 32), (v8f){});  // B[k=a][n=d]=xpT[d][a]
    int l = tid & 31, col = nt*16 + (l & 15), r0 = mt*16 + ((l>>4)<<3);
    #pragma unroll
    for (int i = 0; i < 8; ++i){
      int row = r0 + i;
      atomicAdd(&scores[row], acc[i] * (float)cand_bf[row*128 + col]);
    }
  }
  __syncthreads();

  // ---- scores_I / scores_F (top-k ordering is scale-invariant, sums suffice) ----
  if (tid < 50){
    float si = 0.f, sf = 0.f;
    for (int d = 0; d < 128; ++d){
      float cv = (float)cand_bf[tid*128+d];
      si += cv * uI[d];
      sf += cv * uF[d];
    }
    scoresI[tid] = si; scoresF[tid] = sf;
  }
  __syncthreads();

  // ---- scalar tail: BPR loss, top-4, ssl-contrastive, final atomics ----
  if (tid == 0){
    float pos = scores[0], lsum = 0.f;
    for (int c = 1; c < 50; ++c){
      float z = scores[c] - pos;                      // softplus(neg-pos)
      lsum += fmaxf(z, 0.f) + log1pf(__expf(-fabsf(z)));
    }
    int idxI[4], idxF[4];
    unsigned long long usedI = 0, usedF = 0;
    for (int j = 0; j < 4; ++j){
      float bi = -1e30f, bf = -1e30f; int pi = 0, pf = 0;
      for (int c = 0; c < 50; ++c){
        if (!((usedI >> c) & 1ull) && scoresI[c] > bi){ bi = scoresI[c]; pi = c; }
        if (!((usedF >> c) & 1ull) && scoresF[c] > bf){ bf = scoresF[c]; pf = c; }
      }
      idxI[j] = pi; usedI |= 1ull << pi;
      idxF[j] = pf; usedF |= 1ull << pf;
    }
    float nI = 1e-20f, nF = 1e-20f;
    for (int d = 0; d < 128; ++d){ nI += uI[d]*uI[d]; nF += uF[d]*uF[d]; }
    nI = sqrtf(nI); nF = sqrtf(nF);
    float cs[8];
    for (int j = 0; j < 4; ++j){
      int cF = idxF[j], cI = idxI[j];
      float dF = 0.f, nrF = 1e-20f, dI = 0.f, nrI = 1e-20f;
      for (int d = 0; d < 128; ++d){
        float vF = (float)cand_bf[cF*128+d];
        float vI = (float)cand_bf[cI*128+d];
        dF += vF * uI[d]; nrF += vF*vF;
        dI += vI * uF[d]; nrI += vI*vI;
      }
      cs[j]   = dF / (sqrtf(nrF) * nI);
      cs[4+j] = dI / (sqrtf(nrI) * nF);
    }
    float ps1 = __expf(cs[0]/TEMP_) + __expf(cs[1]/TEMP_);
    float ns1 = __expf(cs[2]/TEMP_) + __expf(cs[3]/TEMP_);
    float ps2 = __expf(cs[4]/TEMP_) + __expf(cs[5]/TEMP_);
    float ns2 = __expf(cs[6]/TEMP_) + __expf(cs[7]/TEMP_);
    float con = log1pf(ns1/ps1) + log1pf(ns2/ps2);   // = log(ps+ns)-log(ps)

    const float invB = 1.0f / 4096.0f;
    atomicAdd(&out[0], lsum * (invB / 49.0f));
    atomicAdd(&out[1], 0.5f * red[0] * invB);
    atomicAdd(&out[2], -red[1] * (invB / 50.0f));
    atomicAdd(&out[3], con * invB);
  }
}

extern "C" void kernel_launch(void* const* d_in, const int* in_sizes, int n_in,
                              void* d_out, int out_size, void* d_ws, size_t ws_size,
                              hipStream_t stream) {
  const int*   history = (const int*)d_in[0];
  const int*   timestp = (const int*)d_in[1];
  const float* adj     = (const float*)d_in[2];
  const int*   pos_id  = (const int*)d_in[3];
  const int*   neg_id  = (const int*)d_in[4];
  const int*   stp     = (const int*)d_in[5];
  const float* emb     = (const float*)d_in[6];
  const float* W1      = (const float*)d_in[7];
  const float* b1      = (const float*)d_in[8];
  const float* W2      = (const float*)d_in[9];
  const float* b2      = (const float*)d_in[10];
  float* out = (float*)d_out;

  duv_zero4<<<1, 32, 0, stream>>>(out);
  duv_block_kernel<<<4096, 256, 0, stream>>>(history, timestp, adj, pos_id,
                                             neg_id, stp, emb, W1, b1, W2, b2, out);
}
